// GraphSAGELinkPredictor_14912126451762
// MI455X (gfx1250) — matrix operand order repaired and checked by reading, not compiled, not run on hardware
//
#include <hip/hip_runtime.h>
#include <cstdint>
#include <cstddef>

// ---------------------------------------------------------------------------
// GraphSAGE link predictor, MI455X (gfx1250, wave32).
// Dense math: v_wmma_f32_16x16x32_bf16 (f32 accumulate).
// Sparse aggregation: wave-per-edge coalesced global_atomic_add_f32 (L2-resident).
// Workspace layout (~156 MB): h, xp, msum (N*128 f32 each) + small buffers +
// bf16 transposed weights. hidden reuses msum after layer 2.
// ---------------------------------------------------------------------------

#define HDIM 128

typedef __attribute__((ext_vector_type(16))) __bf16 v16bf;
typedef __attribute__((ext_vector_type(8)))  __bf16 v8bf;
typedef __attribute__((ext_vector_type(8)))  float  v8f;

// float -> bf16, round-to-nearest-even, pure integer ops (no cvt dependence)
__device__ __forceinline__ __bf16 f2bf(float f) {
  unsigned int u = __builtin_bit_cast(unsigned int, f);
  u = (u + 0x7FFFu + ((u >> 16) & 1u)) >> 16;
  unsigned short s = (unsigned short)u;
  return __builtin_bit_cast(__bf16, s);
}

// A fragment (16x32 bf16): lane<16 -> row=lane, K = {0..15};
// lane>=16 -> row=lane-16, K = {16..31}.
__device__ __forceinline__ v16bf load_a_frag(const __bf16* tile, int lda, int lane, int kbase) {
  int row = lane & 15;
  int hi  = (lane >> 4) & 1;
  const __bf16* p = tile + row * lda + kbase + hi * 16;
  v8bf lo = *(const v8bf*)(p);
  v8bf up = *(const v8bf*)(p + 8);
  return __builtin_shufflevector(lo, up, 0,1,2,3,4,5,6,7,8,9,10,11,12,13,14,15);
}

// B fragment (32x16 bf16): lane holds column N=lane&15, K = e + hi*16 (16 contiguous).
// Weights are stored pre-transposed: Wt[outCol][K], row stride ldk (32B aligned).
__device__ __forceinline__ v16bf load_b_frag(const __bf16* wt, int ldk, int col0, int lane, int kbase) {
  int n  = col0 + (lane & 15);
  int hi = (lane >> 4) & 1;
  return *(const v16bf*)(wt + (size_t)n * ldk + kbase + hi * 16);
}

__device__ __forceinline__ v8f wmma_bf16(v16bf a, v16bf b, v8f c) {
  return __builtin_amdgcn_wmma_f32_16x16x32_bf16(false, a, false, b, (short)0, c, false, false);
}

// ---------------------------------------------------------------------------
// Utility kernels
// ---------------------------------------------------------------------------
__global__ void fill_f32(float* p, float v, long long n) {
  long long i = (long long)blockIdx.x * blockDim.x + threadIdx.x;
  if (i < n) p[i] = v;
}

// W [K][Nout] f32 row-major  ->  Wt [Nout][K] bf16
__global__ void conv_wt(const float* __restrict__ W, __bf16* __restrict__ Wt, int K, int Nout) {
  int i = blockIdx.x * blockDim.x + threadIdx.x;
  if (i >= K * Nout) return;
  int k = i / Nout, n = i - k * Nout;
  Wt[(size_t)n * K + k] = f2bf(W[i]);
}

__global__ void count_k(const int* __restrict__ dst, float* cnt, int e) {
  int i = blockIdx.x * blockDim.x + threadIdx.x;
  if (i < e) atomicAdd(&cnt[dst[i]], 1.0f);
}

__global__ void rcnt_k(const float* __restrict__ cnt, float* __restrict__ rcnt, int n) {
  int i = blockIdx.x * blockDim.x + threadIdx.x;
  if (i < n) rcnt[i] = 1.0f / fmaxf(cnt[i], 1.0f);
}

// ---------------------------------------------------------------------------
// Layer 0 (IN = 2): scalar path
// ---------------------------------------------------------------------------
__global__ void xp0_k(const float* __restrict__ x, const float* __restrict__ Wp,
                      const float* __restrict__ bp, float* __restrict__ xp0, int n) {
  int i = blockIdx.x * blockDim.x + threadIdx.x;
  if (i >= n) return;
  float x0 = x[2 * i], x1 = x[2 * i + 1];
  float a = x0 * Wp[0] + x1 * Wp[2] + bp[0];
  float b = x0 * Wp[1] + x1 * Wp[3] + bp[1];
  xp0[2 * i]     = a > 0.f ? a : 0.f;
  xp0[2 * i + 1] = b > 0.f ? b : 0.f;
}

__global__ void scatter0_k(const float* __restrict__ xp0, const int* __restrict__ src,
                           const int* __restrict__ dst, float* msum0, int e) {
  int i = blockIdx.x * blockDim.x + threadIdx.x;
  if (i >= e) return;
  int s = src[i], d = dst[i];
  atomicAdd(&msum0[2 * d],     xp0[2 * s]);
  atomicAdd(&msum0[2 * d + 1], xp0[2 * s + 1]);
}

// h = m0@Wl(2x128) + bl + xp0@Wr(2x128), fused LayerNorm. One node per block.
__global__ __launch_bounds__(128) void h0_k(const float* __restrict__ msum0,
    const float* __restrict__ rcnt, const float* __restrict__ xp0,
    const float* __restrict__ Wl, const float* __restrict__ bl, const float* __restrict__ Wr,
    const float* __restrict__ lnw, const float* __restrict__ lnb, float* __restrict__ h) {
  __shared__ float red[128];
  int nid = blockIdx.x, j = threadIdx.x;
  float rc = rcnt[nid];
  float m0 = msum0[2 * nid] * rc, m1 = msum0[2 * nid + 1] * rc;
  float p0 = xp0[2 * nid], p1 = xp0[2 * nid + 1];
  float v = m0 * Wl[j] + m1 * Wl[HDIM + j] + bl[j] + p0 * Wr[j] + p1 * Wr[HDIM + j];
  red[j] = v; __syncthreads();
  for (int s = 64; s > 0; s >>= 1) { if (j < s) red[j] += red[j + s]; __syncthreads(); }
  float mu = red[0] * (1.f / 128.f); __syncthreads();
  float d = v - mu;
  red[j] = d * d; __syncthreads();
  for (int s = 64; s > 0; s >>= 1) { if (j < s) red[j] += red[j + s]; __syncthreads(); }
  float var = red[0] * (1.f / 128.f);
  h[(size_t)nid * HDIM + j] = d * rsqrtf(var + 1e-5f) * lnw[j] + lnb[j];
}

// ---------------------------------------------------------------------------
// WMMA GEMM: xp = relu(h @ Wp + bp).  16 rows/block, 8 waves x 16 cols, K=128.
// ---------------------------------------------------------------------------
__global__ __launch_bounds__(256) void gemm_xp_k(const float* __restrict__ A,
    const __bf16* __restrict__ Bt, const float* __restrict__ bias,
    float* __restrict__ out, int nrows) {
  __shared__ __align__(16) __bf16 As[16 * 136];
  int m0 = blockIdx.x * 16;
  for (int idx = threadIdx.x; idx < 16 * HDIM; idx += 256) {
    int r = idx >> 7, c = idx & 127;
    int node = m0 + r; if (node >= nrows) node = nrows - 1;
    As[r * 136 + c] = f2bf(A[(size_t)node * HDIM + c]);
  }
  __syncthreads();
  int lane = threadIdx.x & 31, col0 = (threadIdx.x >> 5) * 16;
  v8f acc = {};
#pragma unroll
  for (int kk = 0; kk < 4; kk++) {
    v16bf a = load_a_frag(As, 136, lane, kk * 32);
    v16bf b = load_b_frag(Bt, HDIM, col0, lane, kk * 32);
    acc = wmma_bf16(a, b, acc);
  }
  int n = col0 + (lane & 15), hi = (lane >> 4) & 1;
  float bv = bias[n];
#pragma unroll
  for (int r = 0; r < 8; r++) {
    int mrow = m0 + r + hi * 8;
    if (mrow < nrows) {
      float v = acc[r] + bv;
      out[(size_t)mrow * HDIM + n] = v > 0.f ? v : 0.f;
    }
  }
}

// Wave-per-edge mean-aggregation scatter: lane i handles features i,i+32,i+64,i+96.
__global__ __launch_bounds__(256) void scatter_h_k(const float* __restrict__ xp,
    const int* __restrict__ src, const int* __restrict__ dst, float* msum, int e) {
  int wid  = (int)((blockIdx.x * (unsigned)blockDim.x + threadIdx.x) >> 5);
  int lane = threadIdx.x & 31;
  if (wid >= e) return;
  int s = src[wid], d = dst[wid];
  const float* xs = xp + (size_t)s * HDIM;
  float* md = msum + (size_t)d * HDIM;
#pragma unroll
  for (int i = 0; i < 4; i++) atomicAdd(&md[lane + 32 * i], xs[lane + 32 * i]);
}

// h = (msum*rcnt)@Wl + xp@Wr + bl.  Dual-accumulator WMMA (8 wmma/wave).
__global__ __launch_bounds__(256) void gemm_out_k(const float* __restrict__ msum,
    const float* __restrict__ rcnt, const float* __restrict__ xp,
    const __bf16* __restrict__ Wlt, const __bf16* __restrict__ Wrt,
    const float* __restrict__ bl, float* __restrict__ h, int nrows) {
  __shared__ __align__(16) __bf16 Am[16 * 136];
  __shared__ __align__(16) __bf16 Ax[16 * 136];
  int m0 = blockIdx.x * 16;
  for (int idx = threadIdx.x; idx < 16 * HDIM; idx += 256) {
    int r = idx >> 7, c = idx & 127;
    int node = m0 + r; if (node >= nrows) node = nrows - 1;
    float rc = rcnt[node];
    Am[r * 136 + c] = f2bf(msum[(size_t)node * HDIM + c] * rc);
    Ax[r * 136 + c] = f2bf(xp[(size_t)node * HDIM + c]);
  }
  __syncthreads();
  int lane = threadIdx.x & 31, col0 = (threadIdx.x >> 5) * 16;
  v8f accL = {}, accR = {};
#pragma unroll
  for (int kk = 0; kk < 4; kk++) {
    v16bf am = load_a_frag(Am, 136, lane, kk * 32);
    v16bf ax = load_a_frag(Ax, 136, lane, kk * 32);
    v16bf bL = load_b_frag(Wlt, HDIM, col0, lane, kk * 32);
    v16bf bR = load_b_frag(Wrt, HDIM, col0, lane, kk * 32);
    accL = wmma_bf16(am, bL, accL);
    accR = wmma_bf16(ax, bR, accR);
  }
  int n = col0 + (lane & 15), hi = (lane >> 4) & 1;
  float bv = bl[n];
#pragma unroll
  for (int r = 0; r < 8; r++) {
    int mrow = m0 + r + hi * 8;
    if (mrow < nrows) h[(size_t)mrow * HDIM + n] = accL[r] + accR[r] + bv;
  }
}

__global__ __launch_bounds__(128) void layernorm_k(float* __restrict__ h,
    const float* __restrict__ w, const float* __restrict__ b) {
  __shared__ float red[128];
  int nid = blockIdx.x, j = threadIdx.x;
  float v = h[(size_t)nid * HDIM + j];
  red[j] = v; __syncthreads();
  for (int s = 64; s > 0; s >>= 1) { if (j < s) red[j] += red[j + s]; __syncthreads(); }
  float mu = red[0] * (1.f / 128.f); __syncthreads();
  float d = v - mu;
  red[j] = d * d; __syncthreads();
  for (int s = 64; s > 0; s >>= 1) { if (j < s) red[j] += red[j + s]; __syncthreads(); }
  float var = red[0] * (1.f / 128.f);
  h[(size_t)nid * HDIM + j] = d * rsqrtf(var + 1e-5f) * w[j] + b[j];
}

// ---------------------------------------------------------------------------
// Edge MLP hidden = relu(concat(h[src],h[dst]) @ W1 + b1).  Gather fused into
// the LDS A-tile; K = 256 (8 WMMA k-chunks).
// ---------------------------------------------------------------------------
__global__ __launch_bounds__(256) void gemm_edge_k(const float* __restrict__ h,
    const int* __restrict__ qsrc, const int* __restrict__ qdst,
    const __bf16* __restrict__ W1t /*[128][256]*/, const float* __restrict__ b1,
    float* __restrict__ hidden, int q) {
  __shared__ __align__(16) __bf16 As[16 * 264];
  int q0 = blockIdx.x * 16;
  for (int idx = threadIdx.x; idx < 16 * 256; idx += 256) {
    int r = idx >> 8, c = idx & 255;
    int qq = q0 + r; if (qq >= q) qq = q - 1;
    int node = (c < HDIM) ? qsrc[qq] : qdst[qq];
    As[r * 264 + c] = f2bf(h[(size_t)node * HDIM + (c & 127)]);
  }
  __syncthreads();
  int lane = threadIdx.x & 31, col0 = (threadIdx.x >> 5) * 16;
  v8f acc = {};
#pragma unroll
  for (int kk = 0; kk < 8; kk++) {
    v16bf a = load_a_frag(As, 264, lane, kk * 32);
    v16bf b = load_b_frag(W1t, 256, col0, lane, kk * 32);
    acc = wmma_bf16(a, b, acc);
  }
  int n = col0 + (lane & 15), hi = (lane >> 4) & 1;
  float bv = b1[n];
#pragma unroll
  for (int r = 0; r < 8; r++) {
    int qrow = q0 + r + hi * 8;
    if (qrow < q) {
      float v = acc[r] + bv;
      hidden[(size_t)qrow * HDIM + n] = v > 0.f ? v : 0.f;
    }
  }
}

__global__ __launch_bounds__(128) void edge_out_k(const float* __restrict__ hidden,
    const float* __restrict__ W2, const float* __restrict__ b2, float* __restrict__ out) {
  __shared__ float red[128];
  int qq = blockIdx.x, j = threadIdx.x;
  red[j] = hidden[(size_t)qq * HDIM + j] * W2[j];
  __syncthreads();
  for (int s = 64; s > 0; s >>= 1) { if (j < s) red[j] += red[j + s]; __syncthreads(); }
  if (j == 0) out[qq] = red[0] + b2[0];
}

// ---------------------------------------------------------------------------
extern "C" void kernel_launch(void* const* d_in, const int* in_sizes, int n_in,
                              void* d_out, int out_size, void* d_ws, size_t ws_size,
                              hipStream_t stream) {
  const float* x      = (const float*)d_in[0];
  const int*   ei     = (const int*)d_in[1];
  const int*   eli    = (const int*)d_in[2];
  const float* p0_Wp  = (const float*)d_in[3];
  const float* p0_bp  = (const float*)d_in[4];
  const float* p0_Wl  = (const float*)d_in[5];
  const float* p0_bl  = (const float*)d_in[6];
  const float* p0_Wr  = (const float*)d_in[7];
  const float* Wp_s   = (const float*)d_in[8];
  const float* bp_s   = (const float*)d_in[9];
  const float* Wl_s   = (const float*)d_in[10];
  const float* bl_s   = (const float*)d_in[11];
  const float* Wr_s   = (const float*)d_in[12];
  const float* ln_w   = (const float*)d_in[13];
  const float* ln_b   = (const float*)d_in[14];
  const float* e_W1   = (const float*)d_in[15];
  const float* e_b1   = (const float*)d_in[16];
  const float* e_W2   = (const float*)d_in[17];
  const float* e_b2   = (const float*)d_in[18];

  const int n = in_sizes[0] / 2;   // N nodes (x is N x 2)
  const int e = in_sizes[1] / 2;   // E edges
  const int q = in_sizes[2] / 2;   // Q queries
  const int* src  = ei;
  const int* dst  = ei + e;
  const int* qsrc = eli;
  const int* qdst = eli + q;

  uint8_t* ws = (uint8_t*)d_ws;
  size_t off = 0;
  auto take = [&](size_t bytes) -> void* {
    void* p = ws + off;
    off = (off + bytes + 255) & ~(size_t)255;
    return p;
  };
  float*  h     = (float*)take((size_t)n * HDIM * 4);
  float*  xp    = (float*)take((size_t)n * HDIM * 4);
  float*  msum  = (float*)take((size_t)n * HDIM * 4);   // reused as `hidden` at the end
  float*  xp0   = (float*)take((size_t)n * 2 * 4);
  float*  msum0 = (float*)take((size_t)n * 2 * 4);
  float*  cnt   = (float*)take((size_t)n * 4);
  float*  rcnt  = (float*)take((size_t)n * 4);
  __bf16* Wpt   = (__bf16*)take((size_t)2 * HDIM * HDIM * 2);
  __bf16* Wlt   = (__bf16*)take((size_t)2 * HDIM * HDIM * 2);
  __bf16* Wrt   = (__bf16*)take((size_t)2 * HDIM * HDIM * 2);
  __bf16* eW1t  = (__bf16*)take((size_t)HDIM * 256 * 2);
  float*  hidden = msum;

  const int TB = 256;
  const int gw = (HDIM * HDIM + TB - 1) / TB;

  // Weight transpose + bf16 convert (tiny; all hit L2 afterwards)
  for (int i = 0; i < 2; i++) {
    conv_wt<<<gw, TB, 0, stream>>>(Wp_s + (size_t)i * HDIM * HDIM, Wpt + (size_t)i * HDIM * HDIM, HDIM, HDIM);
    conv_wt<<<gw, TB, 0, stream>>>(Wl_s + (size_t)i * HDIM * HDIM, Wlt + (size_t)i * HDIM * HDIM, HDIM, HDIM);
    conv_wt<<<gw, TB, 0, stream>>>(Wr_s + (size_t)i * HDIM * HDIM, Wrt + (size_t)i * HDIM * HDIM, HDIM, HDIM);
  }
  conv_wt<<<(256 * HDIM + TB - 1) / TB, TB, 0, stream>>>(e_W1, eW1t, 256, HDIM);

  // Degree counts (shared by all 3 layers)
  fill_f32<<<(n + TB - 1) / TB, TB, 0, stream>>>(cnt, 0.f, n);
  count_k<<<(e + TB - 1) / TB, TB, 0, stream>>>(dst, cnt, e);
  rcnt_k<<<(n + TB - 1) / TB, TB, 0, stream>>>(cnt, rcnt, n);

  // ---- Layer 0 (IN=2, scalar) ----
  xp0_k<<<(n + TB - 1) / TB, TB, 0, stream>>>(x, p0_Wp, p0_bp, xp0, n);
  fill_f32<<<((long long)n * 2 + TB - 1) / TB, TB, 0, stream>>>(msum0, 0.f, (long long)n * 2);
  scatter0_k<<<(e + TB - 1) / TB, TB, 0, stream>>>(xp0, src, dst, msum0, e);
  h0_k<<<n, 128, 0, stream>>>(msum0, rcnt, xp0, p0_Wl, p0_bl, p0_Wr, ln_w, ln_b, h);

  // ---- Layers 1..2 (WMMA) ----
  const int gblk = (n + 15) / 16;
  for (int i = 0; i < 2; i++) {
    gemm_xp_k<<<gblk, TB, 0, stream>>>(h, Wpt + (size_t)i * HDIM * HDIM, bp_s + (size_t)i * HDIM, xp, n);
    fill_f32<<<((long long)n * HDIM + TB - 1) / TB, TB, 0, stream>>>(msum, 0.f, (long long)n * HDIM);
    scatter_h_k<<<((long long)e * 32 + TB - 1) / TB, TB, 0, stream>>>(xp, src, dst, msum, e);
    gemm_out_k<<<gblk, TB, 0, stream>>>(msum, rcnt, xp, Wlt + (size_t)i * HDIM * HDIM,
                                        Wrt + (size_t)i * HDIM * HDIM, bl_s + (size_t)i * HDIM, h, n);
    layernorm_k<<<n, 128, 0, stream>>>(h, ln_w + (size_t)(i + 1) * HDIM, ln_b + (size_t)(i + 1) * HDIM);
  }

  // ---- Edge MLP ----
  gemm_edge_k<<<(q + 15) / 16, TB, 0, stream>>>(h, qsrc, qdst, eW1t, e_b1, hidden, q);
  edge_out_k<<<q, 128, 0, stream>>>(hidden, e_W2, e_b2, (float*)d_out);
}
